// BigramLanguageModel_1090921693663
// MI455X (gfx1250) — matrix operand order repaired and checked by reading, not compiled, not run on hardware
//
#include <hip/hip_runtime.h>
#include <hip/hip_bf16.h>
#include <stdint.h>

#define DEV __device__ __forceinline__

// ---------------- model dimensions (fixed by reference) ----------------
constexpr int Bb = 128, Tt = 256, Cc = 384, Hh = 6, Dd = 64, Vv = 65;
constexpr int FF = 4 * Cc;    // 1536
constexpr int QC = 3 * Cc;    // 1152 (packed q|k|v columns)
constexpr int BT = Bb * Tt;   // 32768 tokens
constexpr int VP = 128;       // padded vocab for logits GEMM
constexpr int BH = Bb * Hh;   // 768 attention batches
constexpr float ATT_SCALE = 0.051031036307982884f; // 384^-0.5 (note: C, not D)

enum { FLAG_RELU = 1, FLAG_CAUSAL = 2 };

// ---------------- WMMA types ----------------
typedef __bf16 bhalf;
typedef bhalf v16bf __attribute__((ext_vector_type(16)));
typedef bhalf v2bf  __attribute__((ext_vector_type(2)));
typedef float v8f  __attribute__((ext_vector_type(8)));

union Frag { v16bf v; uint4 u[2]; };

// native f32 -> bf16 (RNE) via hardware convert
DEV unsigned short f2bf(float f) {
  union { bhalf h; unsigned short u; } cv;
  cv.h = (bhalf)f;
  return cv.u;
}
DEV unsigned int f2bf2(float lo, float hi) {
  union { v2bf h; unsigned int u; } cv;
  cv.h[0] = (bhalf)lo;
  cv.h[1] = (bhalf)hi;
  return cv.u;
}

DEV float wred_sum(float v) {
  for (int o = 16; o > 0; o >>= 1) v += __shfl_xor(v, o, 32);
  return v;
}
DEV float wred_max(float v) {
  for (int o = 16; o > 0; o >>= 1) v = fmaxf(v, __shfl_xor(v, o, 32));
  return v;
}

// ----------------------------------------------------------------------
// Tiled bf16 WMMA GEMM: C[m0:m0+128, n0:n0+128] = act( scale*A@B + bias + resid )
// A : f32 [.,K] row-major, lda % 4 == 0, rows m0..m0+127 valid, K % 32 == 0.
// Bt: bf16 "B^T" [N][K] row-major, ldb % 32 == 0; 128 rows starting at n0
//     must be loadable (padded or safely over-readable inside d_ws).
// Store side guarded by N (and bias read). Double-buffered LDS (32 KB), one
// barrier per K-step, global prefetch two K-steps ahead. 256 threads =
// 8 waves (2M x 4N); each wave computes 64x32 via 4x2 tiles of
// v_wmma_f32_16x16x32_bf16 (8 WMMAs per 6 fragment loads per K-step).
// ----------------------------------------------------------------------
DEV void gemm_tile(const float* __restrict__ A, int lda,
                   const unsigned short* __restrict__ Bt, int ldb,
                   float* __restrict__ Cm, int ldc,
                   const float* __restrict__ bias,
                   const float* __restrict__ resid,
                   int N, int K, int m0, int n0, float scale, int flags)
{
  __shared__ unsigned short As[2][128 * 32];  // [buf][row][k] bf16
  __shared__ unsigned short Bs[2][128 * 32];  // [buf][col][k] bf16 (N-major)

  const int tid  = threadIdx.x;
  const int lane = tid & 31;
  const int wave = tid >> 5;
  const int wm   = wave >> 2;   // 0..1  (64 rows each)
  const int wn   = wave & 3;    // 0..3  (32 cols each)

  v8f acc[4][2];
  for (int mi = 0; mi < 4; ++mi)
    for (int ni = 0; ni < 2; ++ni)
      for (int i = 0; i < 8; ++i) acc[mi][ni][i] = 0.0f;

  // A-tile loader coords: 128 rows x 32 k, 16 floats/thread
  const int ar = tid >> 1;            // 0..127
  const int ac = (tid & 1) << 4;      // 0,16
  // B-tile loader coords: 128 rows x 32 k, 16 bf16/thread
  const int tnr = tid >> 1;           // 0..127
  const int tkc = (tid & 1) << 4;     // 0,16

  const int nk = K >> 5;

  auto aAddr = [&](int kk) {
    return A + (size_t)(m0 + ar) * lda + (kk << 5) + ac;
  };
  auto bAddr = [&](int kk) {
    return Bt + (size_t)(n0 + tnr) * ldb + (kk << 5) + tkc;
  };
  auto loadA = [&](int kk, float4& a0, float4& a1, float4& a2, float4& a3) {
    const float* ap = aAddr(kk);
    a0 = *(const float4*)ap;
    a1 = *(const float4*)(ap + 4);
    a2 = *(const float4*)(ap + 8);
    a3 = *(const float4*)(ap + 12);
  };
  auto loadB = [&](int kk, uint4& b0, uint4& b1) {
    const unsigned short* bp = bAddr(kk);
    b0 = *(const uint4*)bp;
    b1 = *(const uint4*)(bp + 8);
  };
  auto storeTiles = [&](int p, const float4& a0, const float4& a1,
                        const float4& a2, const float4& a3,
                        const uint4& b0, const uint4& b1) {
    uint4 u0, u1;
    u0.x = f2bf2(a0.x, a0.y); u0.y = f2bf2(a0.z, a0.w);
    u0.z = f2bf2(a1.x, a1.y); u0.w = f2bf2(a1.z, a1.w);
    u1.x = f2bf2(a2.x, a2.y); u1.y = f2bf2(a2.z, a2.w);
    u1.z = f2bf2(a3.x, a3.y); u1.w = f2bf2(a3.z, a3.w);
    *(uint4*)&As[p][ar * 32 + ac]     = u0;
    *(uint4*)&As[p][ar * 32 + ac + 8] = u1;
    *(uint4*)&Bs[p][tnr * 32 + tkc]     = b0;
    *(uint4*)&Bs[p][tnr * 32 + tkc + 8] = b1;
  };

  // ---- prologue: stage tile 0, prefetch tile 1 ----
  {
    float4 a0, a1, a2, a3; uint4 b0, b1;
    loadA(0, a0, a1, a2, a3);
    loadB(0, b0, b1);
    if (nk > 1) {
      __builtin_prefetch(aAddr(1), 0, 1);
      __builtin_prefetch(bAddr(1), 0, 1);
    }
    storeTiles(0, a0, a1, a2, a3, b0, b1);
  }
  __syncthreads();

  const int r = lane & 15;
  const int g = lane >> 4;

  for (int kk = 0; kk < nk; ++kk) {
    const int p = kk & 1;

    // ---- issue next tile's global loads early (latency hiding) ----
    float4 na0, na1, na2, na3; uint4 nb0, nb1;
    const bool more = (kk + 1) < nk;
    if (more) { loadA(kk + 1, na0, na1, na2, na3); loadB(kk + 1, nb0, nb1); }
    if (kk + 2 < nk) {
      __builtin_prefetch(aAddr(kk + 2), 0, 1);
      __builtin_prefetch(bAddr(kk + 2), 0, 1);
    }

    // ---- fragment loads + WMMA on current buffer ----
    Frag fa[4], fb[2];
    for (int mi = 0; mi < 4; ++mi) {
      // A layout: lane row = lane%16, K = g*16 + [0..15]
      const unsigned short* pA = &As[p][(wm * 64 + mi * 16 + r) * 32 + g * 16];
      fa[mi].u[0] = *(const uint4*)pA;
      fa[mi].u[1] = *(const uint4*)(pA + 8);
    }
    for (int ni = 0; ni < 2; ++ni) {
      // B layout: lane col = lane%16, K = g*16 + [0..15]
      const unsigned short* pB = &Bs[p][(wn * 32 + ni * 16 + r) * 32 + g * 16];
      fb[ni].u[0] = *(const uint4*)pB;
      fb[ni].u[1] = *(const uint4*)(pB + 8);
    }
    for (int mi = 0; mi < 4; ++mi)
      for (int ni = 0; ni < 2; ++ni)
        acc[mi][ni] = __builtin_amdgcn_wmma_f32_16x16x32_bf16(
            false, fa[mi].v, false, fb[ni].v, (short)0, acc[mi][ni], false, false);

    // ---- stage next tile into the other buffer, then one barrier ----
    if (more) {
      storeTiles(p ^ 1, na0, na1, na2, na3, nb0, nb1);
      __syncthreads();
    }
  }

  // ---- epilogue: C/D layout (ISA 7.12.2): col = lane%16, row = vgpr + 8*(lane/16) ----
  for (int mi = 0; mi < 4; ++mi)
    for (int ni = 0; ni < 2; ++ni) {
      const int col = n0 + wn * 32 + ni * 16 + r;
      if (col >= N) continue;
      const int rbase = m0 + wm * 64 + mi * 16 + g * 8;
      const float bia = bias ? bias[col] : 0.0f;
      for (int i = 0; i < 8; ++i) {
        const int row = rbase + i;
        float v = acc[mi][ni][i] * scale;
        if ((flags & FLAG_CAUSAL) && (col > row)) v = -1.0e30f;
        v += bia;
        if (resid) v += resid[(size_t)row * ldc + col];
        if (flags & FLAG_RELU) v = fmaxf(v, 0.0f);
        Cm[(size_t)row * ldc + col] = v;
      }
    }
}

// ---------------- kernels ----------------
__global__ __launch_bounds__(256) void k_gemm(
    const float* A, int lda, const unsigned short* Bt, int ldb, float* Cm, int ldc,
    const float* bias, const float* resid, int N, int K, float scale, int flags)
{
  gemm_tile(A, lda, Bt, ldb, Cm, ldc, bias, resid, N, K,
            blockIdx.y * 128, blockIdx.x * 128, scale, flags);
}

// S[bb] = causal_mask( (Q @ K^T) * scale ),  bb = b*H + h; kt is bf16 [bb][T][D]
__global__ __launch_bounds__(256) void k_attn_scores(const float* qkv,
                                                     const unsigned short* kt, float* S)
{
  const int bb = blockIdx.z;
  const int b = bb / Hh, h = bb % Hh;
  const float* Q = qkv + (size_t)b * Tt * QC + h * Dd;   // [T,D] f32, ld=QC
  const unsigned short* Kp = kt + (size_t)bb * Tt * Dd;  // [T,D] bf16 = (K^T) N-major
  float* Sp = S + (size_t)bb * Tt * Tt;
  gemm_tile(Q, QC, Kp, Dd, Sp, Tt, nullptr, nullptr, Tt, Dd,
            blockIdx.y * 128, blockIdx.x * 128, ATT_SCALE, FLAG_CAUSAL);
}

// O[b,:,h*D:(h+1)*D] = P @ V; vt is bf16 V^T [bb][D][T] (padded for 128-row reads)
__global__ __launch_bounds__(256) void k_attn_out(const float* S,
                                                  const unsigned short* vt, float* O)
{
  const int bb = blockIdx.z;
  const int b = bb / Hh, h = bb % Hh;
  const float* P = S + (size_t)bb * Tt * Tt;             // [T,T] f32
  const unsigned short* Vt = vt + (size_t)bb * Dd * Tt;  // [D,T] bf16
  float* Op = O + (size_t)b * Tt * Cc + h * Dd;          // ld=C
  gemm_tile(P, Tt, Vt, Tt, Op, Cc, nullptr, nullptr, Dd, Tt,
            blockIdx.y * 128, blockIdx.x * 128, 1.0f, 0);
}

__global__ void k_embed(const float* tok, const float* pos, const int* idx, float* x)
{
  const long i = (long)blockIdx.x * blockDim.x + threadIdx.x; // BT*C total
  const int c = (int)(i % Cc);
  const long bt = i / Cc;
  const int t = (int)(bt % Tt);
  x[i] = tok[(size_t)idx[bt] * Cc + c] + pos[(size_t)t * Cc + c];
}

// generic weight prep: f32 [K, Nin] row-major -> bf16 [Nout, K] (rows >= Nin zero)
__global__ void k_wt(const float* in, int K, int Nin, unsigned short* out)
{
  const long i = (long)blockIdx.x * blockDim.x + threadIdx.x; // Nout*K total
  const int n = (int)(i / K);
  const int k = (int)(i % K);
  out[i] = (n < Nin) ? f2bf(in[(size_t)k * Nin + n]) : (unsigned short)0;
}

// fused pack+transpose+convert of Wq/Wk/Wv [H,C,D] -> bf16 [3C][C]
// row = part*C + h*D + d, col = c
__global__ void k_packqkvT(const float* Wq, const float* Wk, const float* Wv,
                           unsigned short* wt)
{
  const long i = (long)blockIdx.x * blockDim.x + threadIdx.x; // 3C*C total
  const int rowq = (int)(i / Cc);   // 0..1151
  const int c = (int)(i % Cc);
  const int part = rowq / Cc;
  const int hd = rowq % Cc;
  const int h = hd / Dd, d = hd % Dd;
  const float* W = (part == 0) ? Wq : (part == 1) ? Wk : Wv;
  wt[i] = f2bf(W[(size_t)h * Cc * Dd + (size_t)c * Dd + d]);
}

// build bf16 K [bb][T][D] and bf16 V^T [bb][D][T] from qkv [B][T][3C]
__global__ void k_kvt(const float* qkv, unsigned short* kt, unsigned short* vt)
{
  const long i = (long)blockIdx.x * blockDim.x + threadIdx.x; // BH*T*D total
  const int bb = (int)(i / (Tt * Dd));
  const int rem = (int)(i % (Tt * Dd));
  const int s = rem / Dd, d = rem % Dd;
  const int b = bb / Hh, h = bb % Hh;
  const float* base = qkv + (size_t)b * Tt * QC + (size_t)s * QC + h * Dd + d;
  kt[i] = f2bf(base[Cc]);                                             // K, coalesced
  vt[(size_t)bb * Dd * Tt + (size_t)d * Tt + s] = f2bf(base[2 * Cc]); // V^T
}

// one wave per row of C=384
__global__ __launch_bounds__(256) void k_ln(const float* x, const float* g,
                                            const float* b, float* y)
{
  const int row = blockIdx.x * 8 + (threadIdx.x >> 5);
  const int lane = threadIdx.x & 31;
  const float* xr = x + (size_t)row * Cc;
  float v[12];
  float s = 0.0f;
  for (int i = 0; i < 12; ++i) { v[i] = xr[lane + i * 32]; s += v[i]; }
  const float m = wred_sum(s) * (1.0f / Cc);
  float s2 = 0.0f;
  for (int i = 0; i < 12; ++i) { float d = v[i] - m; s2 += d * d; }
  const float inv = rsqrtf(wred_sum(s2) * (1.0f / Cc) + 1e-5f);
  float* yr = y + (size_t)row * Cc;
  for (int i = 0; i < 12; ++i) {
    const int c = lane + i * 32;
    yr[c] = (v[i] - m) * inv * g[c] + b[c];
  }
}

// one wave per row of T=256 (in-place)
__global__ __launch_bounds__(256) void k_softmax(float* S)
{
  const long row = (long)blockIdx.x * 8 + (threadIdx.x >> 5);
  const int lane = threadIdx.x & 31;
  float* r = S + row * Tt;
  float v[8];
  float m = -3.0e38f;
  for (int i = 0; i < 8; ++i) { v[i] = r[lane + i * 32]; m = fmaxf(m, v[i]); }
  m = wred_max(m);
  float s = 0.0f;
  for (int i = 0; i < 8; ++i) { v[i] = expf(v[i] - m); s += v[i]; }
  s = wred_sum(s);
  const float inv = 1.0f / s;
  for (int i = 0; i < 8; ++i) r[lane + i * 32] = v[i] * inv;
}

// one wave per row of V=65 logits; accumulate sum of (logZ - logit[target])
__global__ __launch_bounds__(256) void k_ce(const float* logits, const int* tgt, float* acc)
{
  const int row = blockIdx.x * 8 + (threadIdx.x >> 5);
  const int lane = threadIdx.x & 31;
  const float* lp = logits + (size_t)row * Vv;
  const float v0 = lp[lane];
  const float v1 = (lane + 32 < Vv) ? lp[lane + 32] : -3.0e38f;
  const float v2 = (lane == 0) ? lp[64] : -3.0e38f;
  float m = fmaxf(v0, fmaxf(v1, v2));
  m = wred_max(m);
  float s = expf(v0 - m) + expf(v1 - m) + expf(v2 - m);
  s = wred_sum(s);
  if (lane == 0) {
    const float logZ = m + logf(s);
    atomicAdd(acc, logZ - lp[tgt[row]]);
  }
}

__global__ void k_zero(float* p) { *p = 0.0f; }
__global__ void k_final(const float* acc, float* out) { *out = *acc * (1.0f / (float)BT); }

// ---------------- host launcher ----------------
extern "C" void kernel_launch(void* const* d_in, const int* in_sizes, int n_in,
                              void* d_out, int out_size, void* d_ws, size_t ws_size,
                              hipStream_t stream)
{
  (void)in_sizes; (void)n_in; (void)out_size; (void)ws_size;

  // input layout: tok, pos, 6 x {ln1_g,ln1_b,ln2_g,ln2_b,Wq,Wk,Wv,Wp,bp,W1,b1,W2,b2},
  //               fW1, fb1, fW2, fb2, lm_W, lm_b, idx, targets
  const float* tok = (const float*)d_in[0];
  const float* pos = (const float*)d_in[1];
  const float* fW1  = (const float*)d_in[80];
  const float* fb1  = (const float*)d_in[81];
  const float* fW2  = (const float*)d_in[82];
  const float* fb2  = (const float*)d_in[83];
  const float* lm_W = (const float*)d_in[84];
  const float* lm_b = (const float*)d_in[85];
  const int* idx = (const int*)d_in[86];
  const int* tgt = (const int*)d_in[87];
  float* out = (float*)d_out;

  float* x    = (float*)d_ws;                        // [BT, C]
  float* hbuf = x + (size_t)BT * Cc;                 // [BT, C]   (LN out / attn O / final hidden)
  float* big  = hbuf + (size_t)BT * Cc;              // [BT, 4C]  (qkv then FFN mid)
  float* S    = big + (size_t)BT * FF;               // [B*H, T, T]
  float* lacc = S + (size_t)BH * Tt * Tt;            // loss accumulator
  // bf16 region (64B aligned for uint4 loads)
  unsigned short* wt = (unsigned short*)(((uintptr_t)(lacc + 1) + 63) & ~(uintptr_t)63);
  unsigned short* kt = wt + (size_t)FF * Cc;         // [BH][T][D] bf16 keys
  unsigned short* vt = kt + (size_t)BH * Tt * Dd;    // [BH][D][T] bf16 V^T (+pad below)
  // vt needs BH*D*T + 64*T pad for the 128-row over-read on the last batch

  const dim3 blk(256);

  k_embed<<<dim3(((long)BT * Cc) / 256), blk, 0, stream>>>(tok, pos, idx, x);

  for (int l = 0; l < 6; ++l) {
    const float* const* p = (const float* const*)(d_in + 2 + l * 13);
    const float* ln1_g = p[0];  const float* ln1_b = p[1];
    const float* ln2_g = p[2];  const float* ln2_b = p[3];
    const float* Wq = p[4];     const float* Wk = p[5];  const float* Wv = p[6];
    const float* Wp = p[7];     const float* bp = p[8];
    const float* W1 = p[9];     const float* b1 = p[10];
    const float* W2 = p[11];    const float* b2 = p[12];

    k_ln<<<dim3(BT / 8), blk, 0, stream>>>(x, ln1_g, ln1_b, hbuf);
    k_packqkvT<<<dim3((QC * Cc) / 256), blk, 0, stream>>>(Wq, Wk, Wv, wt);
    // qkv = LN1(x) @ Wqkv           [BT, 3C]
    k_gemm<<<dim3(QC / 128, BT / 128), blk, 0, stream>>>(
        hbuf, Cc, wt, Cc, big, QC, nullptr, nullptr, QC, Cc, 1.0f, 0);
    // bf16 K and V^T per (b,h)
    k_kvt<<<dim3(((long)BH * Tt * Dd) / 256), blk, 0, stream>>>(big, kt, vt);
    // S = mask(Q K^T * scale)       [B*H, T, T]
    k_attn_scores<<<dim3(Tt / 128, Tt / 128, BH), blk, 0, stream>>>(big, kt, S);
    k_softmax<<<dim3((BH * Tt) / 8), blk, 0, stream>>>(S);
    // O = P V (concat heads)        -> hbuf [BT, C]
    k_attn_out<<<dim3(1, Tt / 128, BH), blk, 0, stream>>>(S, vt, hbuf);
    // x = x + O @ Wp + bp
    k_wt<<<dim3((Cc * Cc) / 256), blk, 0, stream>>>(Wp, Cc, Cc, wt);
    k_gemm<<<dim3(Cc / 128, BT / 128), blk, 0, stream>>>(
        hbuf, Cc, wt, Cc, x, Cc, bp, x, Cc, Cc, 1.0f, 0);
    // FFN
    k_ln<<<dim3(BT / 8), blk, 0, stream>>>(x, ln2_g, ln2_b, hbuf);
    k_wt<<<dim3((FF * Cc) / 256), blk, 0, stream>>>(W1, Cc, FF, wt);
    k_gemm<<<dim3(FF / 128, BT / 128), blk, 0, stream>>>(
        hbuf, Cc, wt, Cc, big, FF, b1, nullptr, FF, Cc, 1.0f, FLAG_RELU);
    k_wt<<<dim3((Cc * FF) / 256), blk, 0, stream>>>(W2, FF, Cc, wt);
    k_gemm<<<dim3(Cc / 128, BT / 128), blk, 0, stream>>>(
        big, FF, wt, FF, x, Cc, b2, x, Cc, FF, 1.0f, 0);
  }

  // final FFN (no residual): hbuf = relu(x@fW1+fb1)@fW2+fb2
  k_wt<<<dim3((FF * Cc) / 256), blk, 0, stream>>>(fW1, Cc, FF, wt);
  k_gemm<<<dim3(FF / 128, BT / 128), blk, 0, stream>>>(
      x, Cc, wt, Cc, big, FF, fb1, nullptr, FF, Cc, 1.0f, FLAG_RELU);
  k_wt<<<dim3((Cc * FF) / 256), blk, 0, stream>>>(fW2, FF, Cc, wt);
  k_gemm<<<dim3(Cc / 128, BT / 128), blk, 0, stream>>>(
      big, FF, wt, FF, hbuf, Cc, fb2, nullptr, Cc, FF, 1.0f, 0);

  // logits: lm_W [C,65] -> bf16 [128][C] (rows 65..127 zero), then GEMM -> d_out
  k_wt<<<dim3((VP * Cc) / 256), blk, 0, stream>>>(lm_W, Cc, Vv, wt);
  k_gemm<<<dim3(1, BT / 128), blk, 0, stream>>>(
      hbuf, Cc, wt, Cc, out, Vv, lm_b, nullptr, Vv, Cc, 1.0f, 0);

  // loss -> d_out[BT*V]
  k_zero<<<1, 1, 0, stream>>>(lacc);
  k_ce<<<dim3(BT / 8), blk, 0, stream>>>(out, tgt, lacc);
  k_final<<<1, 1, 0, stream>>>(lacc, out + (size_t)BT * Vv);
}
